// ProteinFeatures_8452495638638
// MI455X (gfx1250) — matrix operand order repaired and checked by reading, not compiled, not run on hardware
//
#include <hip/hip_runtime.h>
#include <hip/hip_bf16.h>
#include <stdint.h>

#define NRES   8192
#define KNB    48
#define EF     128
#define FEAT   416
#define SA_STR 432      // f16 stride of feature tile (multiple of 8, 16B-aligned rows)
#define A2_STR 136      // f16 stride of normalized tile
#define BM     64       // edges per block
#define W2_OFF 8704     // half-offset of staged W2 inside sA (= BM*A2_STR)

typedef __attribute__((ext_vector_type(16))) _Float16 v16h;
typedef __attribute__((ext_vector_type(8)))  _Float16 v8h;
typedef __attribute__((ext_vector_type(8)))  float    v8f;

__device__ __forceinline__ v16h cat8(v8h lo, v8h hi) {
    v16h r;
#pragma unroll
    for (int i = 0; i < 8; ++i) { r[i] = lo[i]; r[i + 8] = hi[i]; }
    return r;
}

__device__ __forceinline__ v8f wmma_f16(v16h a, v16h b, v8f c) {
    // D = A(16x32 f16) * B(32x16 f16) + C(16x16 f32)
    return __builtin_amdgcn_wmma_f32_16x16x32_f16(false, a, false, b, (short)0, c, false, false);
}

__device__ __forceinline__ v8f zero8() {
    v8f z;
#pragma unroll
    for (int q = 0; q < 8; ++q) z[q] = 0.0f;
    return z;
}

// B fragment: 32x16 f16, lane n<16 -> col n, K=k0..k0+7,k0+16..k0+23; lane n+16 -> K=k0+8..k0+15,k0+24..k0+31
__device__ __forceinline__ v16h loadB(const _Float16* base, int col, int stride,
                                      int kc, int lhi) {
    const _Float16* bp = base + (size_t)col * stride + kc * 32 + lhi * 8;
    return cat8(*(const v8h*)bp, *(const v8h*)(bp + 16));
}

__device__ __forceinline__ unsigned long long shflxor_u64(unsigned long long x, int m) {
    unsigned lo = (unsigned)x, hi = (unsigned)(x >> 32);
    lo = __shfl_xor(lo, m, 32);
    hi = __shfl_xor(hi, m, 32);
    return ((unsigned long long)hi << 32) | lo;
}

// ---------------------------------------------------------------------------
// Kernel 1: noisy coords -> 5-atom backbone (N, CA, C, O, virtual CB)
// ---------------------------------------------------------------------------
__global__ void k_bb(const float* __restrict__ coords, const float* __restrict__ noise,
                     const float* __restrict__ bn, float* __restrict__ bb) {
    int i = blockIdx.x * blockDim.x + threadIdx.x;
    if (i >= NRES) return;
    float s = bn[0];
    float a[4][3];
#pragma unroll
    for (int at = 0; at < 4; ++at)
#pragma unroll
        for (int d = 0; d < 3; ++d) {
            int idx = (i * 4 + at) * 3 + d;
            a[at][d] = coords[idx] + s * noise[idx];
        }
    float b[3], cc[3], ax[3];
#pragma unroll
    for (int d = 0; d < 3; ++d) { b[d] = a[1][d] - a[0][d]; cc[d] = a[2][d] - a[1][d]; }
    ax[0] = b[1] * cc[2] - b[2] * cc[1];
    ax[1] = b[2] * cc[0] - b[0] * cc[2];
    ax[2] = b[0] * cc[1] - b[1] * cc[0];
    float* o = bb + (size_t)i * 15;
#pragma unroll
    for (int d = 0; d < 3; ++d) {
        o[0 * 3 + d] = a[0][d];
        o[1 * 3 + d] = a[1][d];
        o[2 * 3 + d] = a[2][d];
        o[3 * 3 + d] = a[3][d];
        o[4 * 3 + d] = -0.58273431f * ax[d] + 0.56802827f * b[d] - 0.54067466f * cc[d] + a[1][d];
    }
}

// ---------------------------------------------------------------------------
// Kernel 2: fp32 -> f16 weight conversion (W1: 128x416, W2: 128x128)
// ---------------------------------------------------------------------------
__global__ void k_cvt(const float* __restrict__ w1, const float* __restrict__ w2,
                      _Float16* __restrict__ W1h, _Float16* __restrict__ W2h) {
    int i = blockIdx.x * 256 + threadIdx.x;
    if (i < EF * FEAT) W1h[i] = (_Float16)w1[i];
    if (i < EF * EF)   W2h[i] = (_Float16)w2[i];
}

// ---------------------------------------------------------------------------
// Kernel 3: KNN (one block per residue). d^2 table in LDS; 48 argmin rounds
// with packed (d2bits<<13 | idx) keys (tie-break: lowest index, like top_k).
// ---------------------------------------------------------------------------
__global__ void __launch_bounds__(256) k_knn(const float* __restrict__ bb, int* __restrict__ nbr) {
    __shared__ float sd[NRES];
    __shared__ unsigned long long warpmin[8];
    int i = blockIdx.x;
    int tid = threadIdx.x;
    int lane = tid & 31, wid = tid >> 5;

    float cx = bb[(size_t)i * 15 + 3];
    float cy = bb[(size_t)i * 15 + 4];
    float cz = bb[(size_t)i * 15 + 5];
    for (int j = tid; j < NRES; j += 256) {
        float dx = cx - bb[(size_t)j * 15 + 3];
        float dy = cy - bb[(size_t)j * 15 + 4];
        float dz = cz - bb[(size_t)j * 15 + 5];
        sd[j] = fmaxf(dx * dx + dy * dy + dz * dz, 0.0f);
    }
    __syncthreads();

    for (int r = 0; r < KNB; ++r) {
        unsigned long long best = ~0ull;
        for (int j = tid; j < NRES; j += 256) {
            unsigned long long key =
                ((unsigned long long)__float_as_uint(sd[j]) << 13) | (unsigned)j;
            best = key < best ? key : best;
        }
#pragma unroll
        for (int off = 16; off; off >>= 1) {
            unsigned long long o = shflxor_u64(best, off);
            best = o < best ? o : best;
        }
        if (lane == 0) warpmin[wid] = best;
        __syncthreads();
        if (wid == 0) {
            unsigned long long b2 = (lane < 8) ? warpmin[lane] : ~0ull;
#pragma unroll
            for (int off = 4; off; off >>= 1) {
                unsigned long long o = shflxor_u64(b2, off);
                b2 = o < b2 ? o : b2;
            }
            if (lane == 0) {
                int w = (int)(b2 & 0x1FFFull);
                nbr[(size_t)i * KNB + r] = w;
                sd[w] = __int_as_float(0x7F800000);  // +inf: exclude
            }
        }
        __syncthreads();
    }
}

// ---------------------------------------------------------------------------
// Kernel 4: edge features + GEMM1 (f16 WMMA) + LayerNorm + GEMM2 + bias.
// Block = 64 edges, 8 waves. Wave w: M-tile (w&3), N-tile group (w>>2).
// W2 is staged into dead LDS via global_load_async_to_lds (ASYNCcnt path),
// overlapped with the LayerNorm repack.
// ---------------------------------------------------------------------------
__global__ void __launch_bounds__(256) k_edge(
    const float* __restrict__ bb, const int* __restrict__ nbr,
    const int* __restrict__ ri, const int* __restrict__ ci,
    const float* __restrict__ wpw, const float* __restrict__ wpb,
    const _Float16* __restrict__ W1h, const _Float16* __restrict__ W2h,
    const float* __restrict__ lnw, const float* __restrict__ lnb,
    const float* __restrict__ wprb, float* __restrict__ out) {

    __shared__ _Float16 sA[BM * SA_STR];      // 55296 B; reused as A2 | staged W2
    __shared__ float sStat[2][BM][2];         // per-row {sum, sumsq} x {colgroup 0,1}

    int tid = threadIdx.x;
    long e0 = (long)blockIdx.x * BM;

    // ---- phase 1: build 64 x 416 f16 feature tile (4 threads per edge) ----
    {
        int el = tid >> 2;          // local edge 0..63
        int sub = tid & 3;
        long e = e0 + el;
        int n = (int)(e / KNB), k = (int)(e % KNB);
        int m = nbr[(size_t)n * KNB + k];
        float pa[5][3], pb[5][3];
#pragma unroll
        for (int a = 0; a < 5; ++a)
#pragma unroll
            for (int d = 0; d < 3; ++d) {
                pa[a][d] = bb[(size_t)n * 15 + a * 3 + d];
                pb[a][d] = bb[(size_t)m * 15 + a * 3 + d];
            }
        float dp[25];
#pragma unroll
        for (int a = 0; a < 5; ++a)
#pragma unroll
            for (int b = 0; b < 5; ++b) {
                float dx = pa[a][0] - pb[b][0];
                float dy = pa[a][1] - pb[b][1];
                float dz = pa[a][2] - pb[b][2];
                dp[a * 5 + b] = sqrtf(dx * dx + dy * dy + dz * dz + 1e-6f);
            }
        int off = ri[n] - ri[m];
        int same = (ci[n] == ci[m]);
        int v = off + 32; v = v < 0 ? 0 : (v > 64 ? 64 : v);
        int enc = same ? v : 65;
        for (int f = sub; f < FEAT; f += 4) {
            float val;
            if (f < 16) {
                val = wpw[f * 66 + enc] + wpb[f];
            } else {
                int j = f - 16, pair = j >> 4, bin = j & 15;
                float mu = 2.0f + (20.0f / 15.0f) * (float)bin;
                float t = (dp[pair] - mu) * 0.8f;   // / sigma (=1.25)
                val = expf(-t * t);
            }
            sA[el * SA_STR + f] = (_Float16)val;
        }
    }
    __syncthreads();

    // ---- GEMM1: e1(64x128) = feat(64x416,f16) * W1^T, f32 accumulate ----
    int lane = tid & 31, w = tid >> 5;
    int mt = w & 3, g = w >> 2;
    int lrow = lane & 15, lhi = lane >> 4;
    int col0 = (g * 4 + 0) * 16 + lrow;
    int col1 = (g * 4 + 1) * 16 + lrow;
    int col2 = (g * 4 + 2) * 16 + lrow;
    int col3 = (g * 4 + 3) * 16 + lrow;

    v8f acc0 = zero8(), acc1 = zero8(), acc2 = zero8(), acc3 = zero8();

    for (int kc = 0; kc < FEAT / 32; ++kc) {
        const _Float16* ap = &sA[(mt * 16 + lrow) * SA_STR + kc * 32 + lhi * 8];
        v16h afrag = cat8(*(const v8h*)ap, *(const v8h*)(ap + 16));
        acc0 = wmma_f16(afrag, loadB(W1h, col0, FEAT, kc, lhi), acc0);
        acc1 = wmma_f16(afrag, loadB(W1h, col1, FEAT, kc, lhi), acc1);
        acc2 = wmma_f16(afrag, loadB(W1h, col2, FEAT, kc, lhi), acc2);
        acc3 = wmma_f16(afrag, loadB(W1h, col3, FEAT, kc, lhi), acc3);
    }

    // ---- LayerNorm stats straight from the accumulators ----
#pragma unroll
    for (int r = 0; r < 8; ++r) {
        float x0 = acc0[r], x1 = acc1[r], x2 = acc2[r], x3 = acc3[r];
        float s = x0 + x1 + x2 + x3;
        float q = x0 * x0 + x1 * x1 + x2 * x2 + x3 * x3;
#pragma unroll
        for (int off = 1; off < 16; off <<= 1) {  // half-wave reduce (rows differ per half)
            s += __shfl_xor(s, off, 32);
            q += __shfl_xor(q, off, 32);
        }
        if (lrow == 0) {
            int row = mt * 16 + r + lhi * 8;
            sStat[0][row][g] = s;
            sStat[1][row][g] = q;
        }
    }
    __syncthreads();   // all GEMM1 LDS reads + stat writes done; sA now re-usable

    // ---- stage W2 (32 KB) into dead LDS via async copy (ASYNCcnt path) ----
    {
        unsigned ldsBase = (unsigned)(size_t)(const void*)(sA + W2_OFF);
        const char* gsrc = (const char*)W2h;
#pragma unroll
        for (int c = 0; c < 8; ++c) {
            unsigned off = ((unsigned)tid + (unsigned)c * 256u) * 16u;   // bytes
            unsigned ldsa = ldsBase + off;
            unsigned long long ga = (unsigned long long)(size_t)(gsrc + off);
            asm volatile("global_load_async_to_lds_b128 %0, %1, off"
                         :: "v"(ldsa), "v"(ga) : "memory");
        }
    }

    // ---- normalize, apply ln_w/ln_b, repack f16 A2 (overlaps async copy) ----
    _Float16* sA2 = sA;
#pragma unroll
    for (int r = 0; r < 8; ++r) {
        int row = mt * 16 + r + lhi * 8;
        float mean = (sStat[0][row][0] + sStat[0][row][1]) * (1.0f / 128.0f);
        float var  = (sStat[1][row][0] + sStat[1][row][1]) * (1.0f / 128.0f) - mean * mean;
        float rinv = rsqrtf(var + 1e-5f);
        float y0 = (acc0[r] - mean) * rinv * lnw[col0] + lnb[col0];
        float y1 = (acc1[r] - mean) * rinv * lnw[col1] + lnb[col1];
        float y2 = (acc2[r] - mean) * rinv * lnw[col2] + lnb[col2];
        float y3 = (acc3[r] - mean) * rinv * lnw[col3] + lnb[col3];
        sA2[row * A2_STR + col0] = (_Float16)y0;
        sA2[row * A2_STR + col1] = (_Float16)y1;
        sA2[row * A2_STR + col2] = (_Float16)y2;
        sA2[row * A2_STR + col3] = (_Float16)y3;
    }
    asm volatile("s_wait_asynccnt 0x0" ::: "memory");  // this wave's W2 copies landed
    __syncthreads();                                   // publish A2 + W2 to all waves

    // ---- GEMM2: e2(64x128) = A2(64x128,f16) * W2^T(LDS), + bias, store ----
    const _Float16* sW2 = sA + W2_OFF;
    v8f d0 = zero8(), d1 = zero8(), d2 = zero8(), d3 = zero8();

    for (int kc = 0; kc < EF / 32; ++kc) {
        const _Float16* ap = &sA2[(mt * 16 + lrow) * A2_STR + kc * 32 + lhi * 8];
        v16h afrag = cat8(*(const v8h*)ap, *(const v8h*)(ap + 16));
        d0 = wmma_f16(afrag, loadB(sW2, col0, EF, kc, lhi), d0);
        d1 = wmma_f16(afrag, loadB(sW2, col1, EF, kc, lhi), d1);
        d2 = wmma_f16(afrag, loadB(sW2, col2, EF, kc, lhi), d2);
        d3 = wmma_f16(afrag, loadB(sW2, col3, EF, kc, lhi), d3);
    }

    float b0 = wprb[col0], b1 = wprb[col1], b2 = wprb[col2], b3 = wprb[col3];
#pragma unroll
    for (int r = 0; r < 8; ++r) {
        long row = e0 + mt * 16 + r + lhi * 8;
        out[row * (long)EF + col0] = d0[r] + b0;
        out[row * (long)EF + col1] = d1[r] + b1;
        out[row * (long)EF + col2] = d2[r] + b2;
        out[row * (long)EF + col3] = d3[r] + b3;
    }
}

// ---------------------------------------------------------------------------
extern "C" void kernel_launch(void* const* d_in, const int* in_sizes, int n_in,
                              void* d_out, int out_size, void* d_ws, size_t ws_size,
                              hipStream_t stream) {
    const float* coords = (const float*)d_in[0];
    // d_in[1] = mask (all ones in this problem; pair masking is a no-op)
    const int*   ri     = (const int*)d_in[2];
    const int*   ci     = (const int*)d_in[3];
    const float* cnoise = (const float*)d_in[4];
    const float* bnoise = (const float*)d_in[5];
    const float* wpw    = (const float*)d_in[6];
    const float* wpb    = (const float*)d_in[7];
    const float* wew    = (const float*)d_in[8];
    const float* lnw    = (const float*)d_in[9];
    const float* lnb    = (const float*)d_in[10];
    const float* wpjw   = (const float*)d_in[11];
    const float* wpjb   = (const float*)d_in[12];

    // workspace layout: bb (N*15 f32) | W1h (128*416 f16) | W2h (128*128 f16)
    float*    bb  = (float*)d_ws;
    _Float16* W1h = (_Float16*)((char*)d_ws + (size_t)NRES * 15 * sizeof(float));
    _Float16* W2h = W1h + (size_t)EF * FEAT;

    float* eOut   = (float*)d_out;
    int*   nbrOut = (int*)(eOut + (size_t)NRES * KNB * EF);  // int32 bits in tail

    k_bb <<<(NRES + 255) / 256, 256, 0, stream>>>(coords, cnoise, bnoise, bb);
    k_cvt<<<(EF * FEAT + 255) / 256, 256, 0, stream>>>(wew, wpjw, W1h, W2h);
    k_knn<<<NRES, 256, 0, stream>>>(bb, nbrOut);
    k_edge<<<(NRES * KNB) / BM, 256, 0, stream>>>(bb, nbrOut, ri, ci, wpw, wpb,
                                                  W1h, W2h, lnw, lnb, wpjb, eOut);
}
